// resRNN_80796924773025
// MI455X (gfx1250) — matrix operand order repaired and checked by reading, not compiled, not run on hardware
//
#include <hip/hip_runtime.h>
#include <cstdint>
#include <cstddef>

// ---------------------------------------------------------------------------
// resRNN for MI455X (gfx1250): f16 WMMA GEMMs (f32->f16 via v_cvt_pk_rtz_f16_f32
// while staging to LDS; weight tiles stay L2-resident), double-buffered LDS,
// fused dual-GEMM + bias, LSTM elementwise, projection GEMM with residual.
//
// Shapes: BATCH=4096, INPUT=1024, HU=512, 4*HU=2048, DEPTH=4, STACKS=2.
// ---------------------------------------------------------------------------

typedef __attribute__((ext_vector_type(16))) _Float16 v16h;
typedef __attribute__((ext_vector_type(8)))  float    v8f;

union FragH {
    v16h  v;
    uint4 q[2];
};

// LDS tile: 128 rows x 32 f16 (64B data), padded to 80B (20 uints) per row.
#define LDS_ROW_U 20
#define LDS_BUF_U (128 * LDS_ROW_U)   // 2560 uints per buffer

__device__ __forceinline__ unsigned pack2_f16(float x, float y) {
#if __has_builtin(__builtin_amdgcn_cvt_pkrtz)
    typedef __attribute__((ext_vector_type(2))) __fp16 v2fp16;
    union { v2fp16 v; unsigned u; } c;
    c.v = __builtin_amdgcn_cvt_pkrtz(x, y);   // v_cvt_pk_rtz_f16_f32: 1 VALU op
    return c.u;
#else
    union { _Float16 h[2]; unsigned u; } c;
    c.h[0] = (_Float16)x;
    c.h[1] = (_Float16)y;
    return c.u;
#endif
}

// Stage a 128x32 f32 tile from global into LDS as f16. 256 threads,
// each loads 4 float4 chunks (global_load_b128) and writes 8B to LDS.
__device__ __forceinline__ void load_tile_f16(unsigned* __restrict__ lds,
                                              const float* __restrict__ src,
                                              size_t ld, int tid) {
#pragma unroll
    for (int it = 0; it < 4; ++it) {
        int chunk = it * 256 + tid;      // 0..1023
        int row   = chunk >> 3;          // 128 rows
        int c4    = chunk & 7;           // 8 float4 per row (32 floats)
        const float4 f = *(const float4*)(src + (size_t)row * ld + c4 * 4);
        unsigned* d = lds + row * LDS_ROW_U + c4 * 2;
        d[0] = pack2_f16(f.x, f.y);
        d[1] = pack2_f16(f.z, f.w);
    }
}

// Prefetch one 128x32 f32 tile into L2 (one line-touch per row).
__device__ __forceinline__ void prefetch_tile(const float* __restrict__ src,
                                              size_t ld, int tid) {
    // 128 rows, 256 threads -> threads 0..127 touch one row each.
    if (tid < 128) {
        __builtin_prefetch(src + (size_t)tid * ld, 0, 3);
    }
}

// One K=32 step: each wave does a 32x64 sub-tile = 2x4 v_wmma_f32_16x16x32_f16.
__device__ __forceinline__ void mma_step(const unsigned* __restrict__ la,
                                         const unsigned* __restrict__ lb,
                                         int wave_m, int wave_n, int lane,
                                         v8f acc[2][4]) {
    const int lr = lane & 15;
    // A frag: lane<16 -> K{0..7,16..23} = uints {0..3, 8..11};
    //         lane>=16 -> K{8..15,24..31} = uints {4..7, 12..15}
    const int ah = (lane & 16) ? 4 : 0;
    // B frag: lane<16 -> K{0..7,16..23} = uints {0..3, 8..11};
    //         lane>=16 -> K{8..15,24..31} = uints {4..7, 12..15}
    const int bh = (lane & 16) ? 4 : 0;

    FragH a[2], b[4];
#pragma unroll
    for (int mi = 0; mi < 2; ++mi) {
        const unsigned* p = la + (wave_m * 32 + mi * 16 + lr) * LDS_ROW_U + ah;
        a[mi].q[0] = *(const uint4*)(p);
        a[mi].q[1] = *(const uint4*)(p + 8);
    }
#pragma unroll
    for (int ni = 0; ni < 4; ++ni) {
        const unsigned* p = lb + (wave_n * 64 + ni * 16 + lr) * LDS_ROW_U + bh;
        b[ni].q[0] = *(const uint4*)(p);
        b[ni].q[1] = *(const uint4*)(p + 8);
    }
#pragma unroll
    for (int mi = 0; mi < 2; ++mi)
#pragma unroll
        for (int ni = 0; ni < 4; ++ni)
            acc[mi][ni] = __builtin_amdgcn_wmma_f32_16x16x32_f16(
                false, a[mi].v, false, b[ni].v, (short)0, acc[mi][ni],
                false, false);
}

// Double-buffered mainloop over one operand pair. A is MxK row-major (lda),
// B is NxK row-major (ldb) i.e. we compute A @ B^T. Pointers pre-offset to
// the block's m0/n0 rows.
__device__ __forceinline__ void gemm_mainloop(const float* __restrict__ A, size_t lda,
                                              const float* __restrict__ B, size_t ldb,
                                              int K,
                                              unsigned* __restrict__ ldsA,
                                              unsigned* __restrict__ ldsB,
                                              int tid, int lane, int wave_m, int wave_n,
                                              v8f acc[2][4]) {
    const int nk = K >> 5;
    load_tile_f16(ldsA, A, lda, tid);
    load_tile_f16(ldsB, B, ldb, tid);
    __syncthreads();
    for (int k = 0; k < nk; ++k) {
        const int cur = k & 1;
        const int nxt = cur ^ 1;
        if (k + 2 < nk) {
            prefetch_tile(A + (size_t)(k + 2) * 32, lda, tid);
            prefetch_tile(B + (size_t)(k + 2) * 32, ldb, tid);
        }
        if (k + 1 < nk) {
            load_tile_f16(ldsA + nxt * LDS_BUF_U, A + (size_t)(k + 1) * 32, lda, tid);
            load_tile_f16(ldsB + nxt * LDS_BUF_U, B + (size_t)(k + 1) * 32, ldb, tid);
        }
        mma_step(ldsA + cur * LDS_BUF_U, ldsB + cur * LDS_BUF_U,
                 wave_m, wave_n, lane, acc);
        __syncthreads();
    }
}

// C[M x N] = A1 @ B1^T (+ A2 @ B2^T) (+ bias1[n]) (+ bias2[n]) (+ resid[m,n])
// Workgroup: 256 threads = 8 wave32, tile 128x128. All dims multiples of 128/32.
__global__ void __launch_bounds__(256)
wmma_gemm_kernel(const float* __restrict__ A1, int lda1, int K1,
                 const float* __restrict__ B1, int ldb1,
                 const float* __restrict__ A2, int lda2, int K2,
                 const float* __restrict__ B2, int ldb2,
                 const float* __restrict__ bias1,
                 const float* __restrict__ bias2,
                 const float* __restrict__ resid, int ldr,
                 float* __restrict__ C, int ldc) {
    __shared__ __align__(16) unsigned ldsA[2 * LDS_BUF_U];
    __shared__ __align__(16) unsigned ldsB[2 * LDS_BUF_U];

    const int tid    = threadIdx.x;
    const int lane   = tid & 31;
    const int wave   = tid >> 5;
    const int wave_m = wave & 3;   // 4 waves along M (32 rows each)
    const int wave_n = wave >> 2;  // 2 waves along N (64 cols each)

    const size_t m0 = (size_t)blockIdx.y * 128;
    const size_t n0 = (size_t)blockIdx.x * 128;

    v8f acc[2][4];
#pragma unroll
    for (int mi = 0; mi < 2; ++mi)
#pragma unroll
        for (int ni = 0; ni < 4; ++ni)
            acc[mi][ni] = (v8f){0.f, 0.f, 0.f, 0.f, 0.f, 0.f, 0.f, 0.f};

    gemm_mainloop(A1 + m0 * (size_t)lda1, (size_t)lda1,
                  B1 + n0 * (size_t)ldb1, (size_t)ldb1, K1,
                  ldsA, ldsB, tid, lane, wave_m, wave_n, acc);
    if (A2 != nullptr && K2 > 0) {
        gemm_mainloop(A2 + m0 * (size_t)lda2, (size_t)lda2,
                      B2 + n0 * (size_t)ldb2, (size_t)ldb2, K2,
                      ldsA, ldsB, tid, lane, wave_m, wave_n, acc);
    }

    // Epilogue. C/D layout: VGPR r -> row m0+16*mi+r(+8 if lane>=16), col = n0+16*ni+(lane&15)
#pragma unroll
    for (int ni = 0; ni < 4; ++ni) {
        const size_t col = n0 + wave_n * 64 + ni * 16 + (lane & 15);
        float badd = 0.f;
        if (bias1) badd += bias1[col];
        if (bias2) badd += bias2[col];
#pragma unroll
        for (int mi = 0; mi < 2; ++mi) {
            const size_t rbase = m0 + wave_m * 32 + mi * 16 + ((lane & 16) ? 8 : 0);
#pragma unroll
            for (int r = 0; r < 8; ++r) {
                const size_t row = rbase + r;
                float v = acc[mi][ni][r] + badd;
                if (resid) v += resid[row * (size_t)ldr + col];
                C[row * (size_t)ldc + col] = v;
            }
        }
    }
}

__device__ __forceinline__ float sigmoidf_fast(float x) {
    return 1.0f / (1.0f + __expf(-x));
}

// Elementwise LSTM cell + residual + output scatter.
// gates: [4096, 2048] (i|f|g|o blocks of 512). state: (2,2,4096,4096) f32.
// out: ho (4096*4096) then newState (2,2,4096,4096). hout: h_out [4096,512].
__global__ void __launch_bounds__(256)
lstm_cell_kernel(const float* __restrict__ gates,
                 const float* __restrict__ state,
                 float* __restrict__ out,
                 float* __restrict__ hout,
                 int depth, int stack) {
    const size_t e = (size_t)blockIdx.x * 256 + threadIdx.x;  // 4096*512 total
    const size_t b = e >> 9;
    const int    u = (int)(e & 511);

    const float* g = gates + b * 2048;
    const float ig = sigmoidf_fast(g[u]);
    const float fg = sigmoidf_fast(g[512 + u]);
    const float gg = tanhf(g[1024 + u]);
    const float og = sigmoidf_fast(g[1536 + u]);

    const size_t HB  = (size_t)4096 * 4096;               // one (batch,hidden) plane
    const size_t col = (size_t)depth * 512 + u;
    const size_t idx = (size_t)stack * HB + b * 4096 + col;

    const float hx = state[idx];           // state[0, stack, b, col]
    const float cx = state[2 * HB + idx];  // state[1, stack, b, col]

    const float co = fg * cx + ig * gg;
    const float h  = og * tanhf(co);

    float* ns = out + HB;                  // newState base
    ns[idx]          = h + hx;             // newH[stack, b, col]
    ns[2 * HB + idx] = co;                 // newC[stack, b, col]
    if (stack == 1) out[b * 4096 + col] = h + hx;  // ho = newH[-1]
    hout[b * 512 + u] = h;                 // raw h_out feeds projection
}

extern "C" void kernel_launch(void* const* d_in, const int* in_sizes, int n_in,
                              void* d_out, int out_size, void* d_ws, size_t ws_size,
                              hipStream_t stream) {
    (void)in_sizes; (void)n_in; (void)out_size; (void)ws_size;

    const float* input  = (const float*)d_in[0];  // (4096, 1024)
    const float* state  = (const float*)d_in[1];  // (2, 2, 4096, 4096)
    const float* W_ih   = (const float*)d_in[2];  // (4, 2, 2048, 1024)
    const float* W_hh   = (const float*)d_in[3];  // (4, 2, 2048, 512)
    const float* b_ih   = (const float*)d_in[4];  // (4, 2, 2048)
    const float* b_hh   = (const float*)d_in[5];  // (4, 2, 2048)
    const float* proj_W = (const float*)d_in[6];  // (1024, 512)
    const float* proj_b = (const float*)d_in[7];  // (1024,)
    float* out = (float*)d_out;

    char* ws = (char*)d_ws;
    float* gates = (float*)ws;                                           // 32 MB
    float* hout  = (float*)(ws + (size_t)4096 * 2048 * 4);               //  8 MB
    float* decin = (float*)(ws + (size_t)4096 * 2048 * 4
                               + (size_t)4096 * 512 * 4);                // 16 MB

    const dim3 blk(256);
    const dim3 gGates(2048 / 128, 4096 / 128);  // (16, 32)
    const dim3 gProj(1024 / 128, 4096 / 128);   // (8, 32)
    const dim3 gLstm((4096 * 512) / 256);

    const size_t HB = (size_t)4096 * 4096;

    for (int i = 0; i < 4; ++i) {
        for (int j = 0; j < 2; ++j) {
            const float* A1  = (j == 0) ? input : decin;
            const float* wih = W_ih + (size_t)(i * 2 + j) * 2048 * 1024;
            const float* whh = W_hh + (size_t)(i * 2 + j) * 2048 * 512;
            const float* hx  = state + (size_t)j * HB + (size_t)i * 512;  // state[0,j,:,s:]
            // gates = A1 @ W_ih^T + hx @ W_hh^T + b_ih + b_hh
            wmma_gemm_kernel<<<gGates, blk, 0, stream>>>(
                A1, 1024, 1024, wih, 1024,
                hx, 4096, 512, whh, 512,
                b_ih + (size_t)(i * 2 + j) * 2048,
                b_hh + (size_t)(i * 2 + j) * 2048,
                nullptr, 0, gates, 2048);

            lstm_cell_kernel<<<gLstm, blk, 0, stream>>>(gates, state, out, hout, i, j);

            if (j == 0) {
                // decin = input + h_out @ proj_W^T + proj_b
                wmma_gemm_kernel<<<gProj, blk, 0, stream>>>(
                    hout, 512, 512, proj_W, 512,
                    nullptr, 0, 0, nullptr, 0,
                    proj_b, nullptr, input, 1024, decin, 1024);
            }
        }
    }
}
